// FeatureInteractionLayer_54451595379107
// MI455X (gfx1250) — matrix operand order repaired and checked by reading, not compiled, not run on hardware
//
#include <hip/hip_runtime.h>
#include <hip/hip_bf16.h>
#include <stdint.h>

// ---------------------------------------------------------------------------
// FeatureInteraction: feats(x) @ W + b -> LayerNorm -> ReLU
//   B=4096, d=64, K=63744 (= 64 + C(64,2) + C(64,3)), N=512
// bf16 WMMA GEMM with on-the-fly A generation; W pre-transposed & converted
// to bf16 (65 MB -> resident in 192MB L2). LayerNorm fused per block.
// ---------------------------------------------------------------------------

#define D_IN     64
#define N_OUT    512
#define K_TOT    63744
#define KC       32           // K per WMMA chunk
#define NCHUNKS  (K_TOT / KC) // 1992
#define MT       32           // output rows per block
#define AROW     40           // bf16 elements per A-tile LDS row (80B: 16B aligned)
#define XROW     72           // f32 elements per x LDS row (64 data + sentinel + pad)
#define LN_EPS   1e-5f

typedef __attribute__((ext_vector_type(16))) __bf16 v16bf;
typedef __attribute__((ext_vector_type(8)))  float  v8f;

union AFrag { v16bf v; uint4 u4[2]; };
union BFrag { v16bf v; uint4 u4[2]; };
union CFrag { v8f  v; float f[8]; };

__device__ __forceinline__ uint16_t f2bf(float f) {
    uint32_t u = __float_as_uint(f);
    uint32_t r = u + 0x7FFFu + ((u >> 16) & 1u);   // round-to-nearest-even
    return (uint16_t)(r >> 16);
}

// ---------------------------------------------------------------------------
// Prep 1: unrank lexicographic combination indices into packed bytes.
// idx[t] = i | j<<8 | k<<16 ; absent slots use 64 (x LDS row has xs[64]=1.0f).
// ---------------------------------------------------------------------------
__global__ void fi_build_idx(uint32_t* __restrict__ idx) {
    int t = blockIdx.x * 256 + threadIdx.x;
    if (t >= K_TOT) return;
    int i, j, k;
    if (t < 64) {                       // linear terms
        i = t; j = 64; k = 64;
    } else if (t < 64 + 2016) {         // pairs i<j
        int p = t - 64; i = 0;
        while (p >= 63 - i) { p -= 63 - i; ++i; }
        j = i + 1 + p; k = 64;
    } else {                            // triples i<j<k
        int tr = t - 2080; i = 0;
        for (;;) { int c = (63 - i) * (62 - i) / 2; if (tr < c) break; tr -= c; ++i; }
        j = i + 1;
        while (tr >= 63 - j) { tr -= 63 - j; ++j; }
        k = j + 1 + tr;
    }
    idx[t] = (uint32_t)i | ((uint32_t)j << 8) | ((uint32_t)k << 16);
}

// ---------------------------------------------------------------------------
// Prep 2: Wt[n][k] = bf16(W[k][n]).  Tiled 64x64 transpose through LDS.
// Grid: (K_TOT/64, N_OUT/64), 256 threads.
// ---------------------------------------------------------------------------
__global__ __launch_bounds__(256) void fi_wt(const float* __restrict__ W,
                                             uint16_t* __restrict__ Wt) {
    __shared__ uint16_t tile[64][72];   // [n][k], padded
    const int kb = blockIdx.x * 64, nb = blockIdx.y * 64;
    const int r  = threadIdx.x >> 2;          // 0..63
    const int c0 = (threadIdx.x & 3) * 16;    // 0,16,32,48
    const float* src = W + (size_t)(kb + r) * N_OUT + nb + c0;
#pragma unroll
    for (int e = 0; e < 16; ++e) tile[c0 + e][r] = f2bf(src[e]);
    __syncthreads();
    uint16_t* dst = Wt + (size_t)(nb + r) * K_TOT + kb + c0;
    uint4 o[2];
    uint16_t* op = (uint16_t*)o;
#pragma unroll
    for (int e = 0; e < 16; ++e) op[e] = tile[r][c0 + e];
    *(uint4*)(dst)     = o[0];
    *(uint4*)(dst + 8) = o[1];
}

// ---------------------------------------------------------------------------
// Main kernel: 128 blocks x 256 threads (8 waves). Block = 32 rows x 512 cols.
// Wave w owns n in [64w, 64w+64): 4 n-slices x 2 row-tiles of 16x16 WMMA acc.
// Per chunk: B loads issued first (hidden behind A-gen + barrier), then gen,
// barrier, A ds_load_b128, 8x WMMA. Prefetch next Wt line (+2 chunks).
// ---------------------------------------------------------------------------
#define SMEM_BYTES (MT * N_OUT * 4 + MT * XROW * 4 + 2 * MT * AROW * 2) // 79872

__global__ __launch_bounds__(256, 1) void fi_main(
    const float* __restrict__ x, const uint32_t* __restrict__ idx,
    const uint16_t* __restrict__ Wt, const float* __restrict__ bias,
    const float* __restrict__ gamma, const float* __restrict__ beta,
    float* __restrict__ out)
{
    extern __shared__ char smem[];
    float*    hsh  = (float*)smem;                                  // [32][512]
    float*    xs   = (float*)(smem + MT * N_OUT * 4);               // [32][72]
    uint16_t* abuf = (uint16_t*)(smem + MT * N_OUT * 4 + MT * XROW * 4); // [2][32][40]

    const int tid  = threadIdx.x;
    const int wave = tid >> 5;
    const int lane = tid & 31;
    const int m0   = blockIdx.x * MT;

    // ---- stage x tile: 32 rows x 64 f32 (+ sentinel 1.0 at col 64) ----
    {
        const int r  = tid >> 3;         // 0..31
        const int c0 = (tid & 7) * 8;    // 0..56
        const float* src = x + (size_t)(m0 + r) * D_IN + c0;
        float4 a = *(const float4*)(src);
        float4 b = *(const float4*)(src + 4);
        float* dst = xs + r * XROW + c0;
        *(float4*)(dst)     = a;
        *(float4*)(dst + 4) = b;
        if ((tid & 7) == 0) xs[r * XROW + 64] = 1.0f;
    }

    // ---- accumulators: [row-tile][n-slice] ----
    CFrag acc[2][4];
#pragma unroll
    for (int rt = 0; rt < 2; ++rt)
#pragma unroll
        for (int s = 0; s < 4; ++s)
#pragma unroll
            for (int v = 0; v < 8; ++v) acc[rt][s].f[v] = 0.0f;

    // ---- per-lane constant addressing ----
    const int ncol  = lane & 15;
    const int khalf = (lane >> 4) * 8;           // B: lanes 16-31 hold K+8 / K+24
    const uint16_t* bbase[4];
#pragma unroll
    for (int s = 0; s < 4; ++s) {
        const int n = wave * 64 + s * 16 + ncol;
        bbase[s] = Wt + (size_t)n * K_TOT + khalf;
    }
    const int arow  = lane & 15;                 // A: row within row-tile
    const int akoff = (lane >> 4) ? 8 : 0;       // A: lanes 16-31 hold K+8 / K+24

    // generation role: row gr, 4 consecutive k starting at gk
    const int gr = tid >> 3;
    const int gk = (tid & 7) * 4;
    const float* xrow = xs + gr * XROW;

    __syncthreads();  // x tile visible to generators

    // =================== K loop: 1992 chunks of 32 ===================
    for (int c = 0; c < NCHUNKS; ++c) {
        const int kb = c * KC;
        uint16_t* ab = abuf + (c & 1) * (MT * AROW);

        // ---- B fragments first: latency hidden behind gen phase + barrier ----
        BFrag bf[4];
#pragma unroll
        for (int s = 0; s < 4; ++s) {
            const uint16_t* bp = bbase[s] + kb;
            bf[s].u4[0] = *(const uint4*)(bp);
            bf[s].u4[1] = *(const uint4*)(bp + 16);
            // pull next 128B line of Wt toward the WGP (covers chunks c+1,c+2)
            __builtin_prefetch(bp + 2 * KC, 0, 0);
        }

        // ---- generate 4 bf16 features for (row gr, k = kb+gk .. +3) ----
        {
            const uint4 t4 = *(const uint4*)(idx + kb + gk);
            const float f0 = xrow[t4.x & 0xff] * xrow[(t4.x >> 8) & 0xff] * xrow[(t4.x >> 16) & 0xff];
            const float f1 = xrow[t4.y & 0xff] * xrow[(t4.y >> 8) & 0xff] * xrow[(t4.y >> 16) & 0xff];
            const float f2 = xrow[t4.z & 0xff] * xrow[(t4.z >> 8) & 0xff] * xrow[(t4.z >> 16) & 0xff];
            const float f3 = xrow[t4.w & 0xff] * xrow[(t4.w >> 8) & 0xff] * xrow[(t4.w >> 16) & 0xff];
            const uint32_t p01 = (uint32_t)f2bf(f0) | ((uint32_t)f2bf(f1) << 16);
            const uint32_t p23 = (uint32_t)f2bf(f2) | ((uint32_t)f2bf(f3) << 16);
            *(uint2*)(ab + gr * AROW + gk) = make_uint2(p01, p23);
        }
        __syncthreads();  // double-buffered: one barrier per chunk suffices

        // ---- A fragments: two ds_read_b128 per row-tile (K-contiguous) ----
        AFrag afr[2];
#pragma unroll
        for (int rt = 0; rt < 2; ++rt) {
            const uint16_t* ap = ab + (arow + rt * 16) * AROW + akoff;
            afr[rt].u4[0] = *(const uint4*)(ap);       // K 0..7  (+akoff)
            afr[rt].u4[1] = *(const uint4*)(ap + 16);  // K 16..23(+akoff)
        }

        // ---- 8x WMMA ----
#pragma unroll
        for (int s = 0; s < 4; ++s) {
#pragma unroll
            for (int rt = 0; rt < 2; ++rt) {
                acc[rt][s].v = __builtin_amdgcn_wmma_f32_16x16x32_bf16(
                    false, afr[rt].v, false, bf[s].v,
                    (short)0, acc[rt][s].v, false, false);
            }
        }
    }

    // ---- epilogue: bias, stage h to LDS ----
#pragma unroll
    for (int s = 0; s < 4; ++s) {
        const int n  = wave * 64 + s * 16 + ncol;
        const float bv = bias[n];
#pragma unroll
        for (int rt = 0; rt < 2; ++rt) {
            const int mbase = rt * 16 + ((lane >> 4) << 3);  // lanes 16-31: M = v+8
#pragma unroll
            for (int v = 0; v < 8; ++v)
                hsh[(mbase + v) * N_OUT + n] = acc[rt][s].f[v] + bv;
        }
    }
    __syncthreads();

    // ---- LayerNorm + ReLU: each wave normalizes 4 rows ----
    for (int rr = 0; rr < 4; ++rr) {
        const int row = wave * 4 + rr;
        const float* hr = hsh + row * N_OUT;
        float s1 = 0.0f, s2 = 0.0f;
#pragma unroll
        for (int jj = 0; jj < 16; ++jj) {
            float v = hr[lane + jj * 32];
            s1 += v; s2 += v * v;
        }
#pragma unroll
        for (int off = 16; off > 0; off >>= 1) {
            s1 += __shfl_xor(s1, off, 32);
            s2 += __shfl_xor(s2, off, 32);
        }
        const float mu  = s1 * (1.0f / N_OUT);
        const float var = s2 * (1.0f / N_OUT) - mu * mu;
        const float rs  = rsqrtf(var + LN_EPS);
        float* orow = out + (size_t)(m0 + row) * N_OUT;
#pragma unroll
        for (int jj = 0; jj < 16; ++jj) {
            const int n = lane + jj * 32;
            const float v = (hr[n] - mu) * rs * gamma[n] + beta[n];
            orow[n] = fmaxf(v, 0.0f);
        }
    }
}

// ---------------------------------------------------------------------------
extern "C" void kernel_launch(void* const* d_in, const int* in_sizes, int n_in,
                              void* d_out, int out_size, void* d_ws, size_t ws_size,
                              hipStream_t stream) {
    const float* x     = (const float*)d_in[0];  // [4096,64]
    const float* W     = (const float*)d_in[1];  // [63744,512]
    const float* bias  = (const float*)d_in[2];  // [512]
    const float* gamma = (const float*)d_in[3];  // [512]
    const float* beta  = (const float*)d_in[4];  // [512]
    float* out = (float*)d_out;                  // [4096,512]

    // workspace: idx table (255 KB) @0, bf16 W-transpose (65.3 MB) @1MB
    uint32_t* idx = (uint32_t*)d_ws;
    uint16_t* Wt  = (uint16_t*)((char*)d_ws + (1u << 20));

    fi_build_idx<<<(K_TOT + 255) / 256, 256, 0, stream>>>(idx);
    fi_wt<<<dim3(K_TOT / 64, N_OUT / 64), 256, 0, stream>>>(W, Wt);
    fi_main<<<4096 / MT, 256, SMEM_BYTES, stream>>>(x, idx, Wt, bias, gamma, beta, out);
}
